// AggCF_Module_74466142978597
// MI455X (gfx1250) — matrix operand order repaired and checked by reading, not compiled, not run on hardware
//
#include <hip/hip_runtime.h>
#include <hip/hip_bf16.h>

// ---------------------------------------------------------------------------
// AggCF module for MI455X (gfx1250). All matmuls via v_wmma_f32_16x16x32_bf16.
// All WMMA operands are arranged contiguous-in-k so every fragment load is
// exactly two ds_load_b128 / global_load_b128 (no scalar ds_load_u16 chains).
//   1) cvt x -> bf16; pool+cvt xp -> bf16; transpose-cvt weights -> bf16
//   2) GEMM (computes C^T = W^T * X^T): Q = x*Wq (pre-scaled log2e/sqrt(dk)),
//      K = xp*Wk, V = xp*Wv (V stored transposed: 512 x 4096)
//   3) flash attention per (b,h): computes S^T = K*Q^T; the S^T register
//      layout doubles as the B-operand layout of O^T = V^T*P^T, so the
//      probability matrix never leaves VGPRs.
//   4) GEMM: out = AO*Wo + b_o + residual(x)  (fp32, vectorized epilogue)
// ---------------------------------------------------------------------------

typedef __attribute__((ext_vector_type(16))) __bf16 v16bf;
typedef __attribute__((ext_vector_type(8)))  float  v8f;

#define DEV __device__ __forceinline__

DEV __bf16 f2bf(float f) {
    unsigned u = __builtin_bit_cast(unsigned, f);
    unsigned r = (u + 0x7FFFu + ((u >> 16) & 1u)) >> 16;
    unsigned short s = (unsigned short)r;
    return __builtin_bit_cast(__bf16, s);
}

DEV v8f wmma_bf16(v16bf a, v16bf b, v8f c) {
    return __builtin_amdgcn_wmma_f32_16x16x32_bf16(false, a, false, b,
                                                   (short)0, c, false, false);
}

// 16-bit operand fragment: per ISA 7.12.2 lane (col/row, half g) holds
// k = {8g..8g+7} then {16+8g..16+8g+7}: two contiguous 16B runs.
union BF16x16 { v16bf v; uint4 q[2]; };
DEV v16bf ldfrag(const __bf16* p, int g) {
    BF16x16 u;
    u.q[0] = *reinterpret_cast<const uint4*>(p + 8 * g);
    u.q[1] = *reinterpret_cast<const uint4*>(p + 16 + 8 * g);
    return u.v;
}

union F8 { float f[8]; float4 q[2]; };
union BF8PK { __bf16 b[8]; uint4 q; };

// ---------------------------------------------------------------------------
// fp32 -> bf16 elementwise convert
// ---------------------------------------------------------------------------
__global__ __launch_bounds__(256) void cvt_bf16_kernel(
    const float* __restrict__ in, __bf16* __restrict__ out, int n) {
    int i = blockIdx.x * 256 + threadIdx.x;
    if (i < n) out[i] = f2bf(in[i]);
}

// fp32 (R x C) -> bf16 transposed (C x R)
__global__ __launch_bounds__(256) void cvt_t_bf16_kernel(
    const float* __restrict__ in, __bf16* __restrict__ out, int R, int C) {
    int i = blockIdx.x * 256 + threadIdx.x;
    if (i >= R * C) return;
    int r = i / C, c = i - r * C;
    out[(size_t)c * R + r] = f2bf(in[i]);
}

// ---------------------------------------------------------------------------
// TF-SAME 3x3/s2 average pool (count-valid divisor), fused fp32->bf16.
// in: (4,64,64,512) fp32, out: (4,32,32,512) bf16. pad_begin = 0.
// ---------------------------------------------------------------------------
__global__ __launch_bounds__(256) void pool_cvt_kernel(
    const float* __restrict__ x, __bf16* __restrict__ xp) {
    int i = blockIdx.x * 256 + threadIdx.x;
    if (i >= 4 * 32 * 32 * 512) return;
    int c = i & 511;
    int w = (i >> 9) & 31;
    int h = (i >> 14) & 31;
    int b = i >> 19;
    float sum = 0.0f;
    int cnt = 0;
    #pragma unroll
    for (int dy = 0; dy < 3; ++dy) {
        int y = h * 2 + dy;
        if (y >= 64) continue;
        #pragma unroll
        for (int dx = 0; dx < 3; ++dx) {
            int xx = w * 2 + dx;
            if (xx >= 64) continue;
            sum += x[(((size_t)b * 64 + y) * 64 + xx) * 512 + c];
            ++cnt;
        }
    }
    xp[i] = f2bf(sum / (float)cnt);
}

// ---------------------------------------------------------------------------
// bf16 WMMA GEMM computing C = A(MxK) * W  with  Wt = W^T (N x K row-major).
// Block tile: 64 rows x 128 channels; wave w owns 32 channels x 64 rows
// (8 accumulators). Per 32-k chunk: 2 A-frags (channels) x 4 B-frags (rows)
// -> 12 ds_load_b128 for 8 WMMAs. Both LDS tiles are row-major-in-k.
// Epilogue: lane holds 8 consecutive output channels -> vectorized
// bias/resid loads and 16B/32B stores.
// transpose_out: store bf16 output as (N x M) instead (for V).
// ---------------------------------------------------------------------------
__global__ __launch_bounds__(128) void gemm_bf16_kernel(
    const __bf16* __restrict__ A, const __bf16* __restrict__ Wt,
    const float* __restrict__ bias, const float* __restrict__ resid,
    __bf16* __restrict__ outb, float* __restrict__ outf,
    int M, int N, int K, float scale, int transpose_out) {
    __shared__ __bf16 Xl[64 * 32];    // [row][k]
    __shared__ __bf16 Wl[128 * 32];   // [channel][k]
    const int lane = threadIdx.x & 31;
    const int wave = threadIdx.x >> 5;
    const int g    = lane >> 4;
    const int r16  = lane & 15;
    const int mblk = blockIdx.y * 64;
    const int nblk = blockIdx.x * 128;

    v8f acc[2][4] = {};

    for (int k0 = 0; k0 < K; k0 += 32) {
        __syncthreads();
        {   // Xl: 2048 elems, 16 per thread (2 x uint4)
            int row = threadIdx.x >> 1, sub = (threadIdx.x & 1) << 4;
            const uint4* sx = reinterpret_cast<const uint4*>(
                A + (size_t)(mblk + row) * K + k0 + sub);
            uint4* dx = reinterpret_cast<uint4*>(Xl + row * 32 + sub);
            dx[0] = sx[0];
            dx[1] = sx[1];
            // Wl: 4096 elems, 32 per thread (full k-row, 4 x uint4)
            const uint4* sw = reinterpret_cast<const uint4*>(
                Wt + (size_t)(nblk + threadIdx.x) * K + k0);
            uint4* dw = reinterpret_cast<uint4*>(Wl + threadIdx.x * 32);
            dw[0] = sw[0];
            dw[1] = sw[1];
            dw[2] = sw[2];
            dw[3] = sw[3];
        }
        __syncthreads();

        v16bf aw0 = ldfrag(Wl + (wave * 32 + r16) * 32, g);
        v16bf aw1 = ldfrag(Wl + (wave * 32 + 16 + r16) * 32, g);
        #pragma unroll
        for (int mt = 0; mt < 4; ++mt) {
            v16bf bx = ldfrag(Xl + (mt * 16 + r16) * 32, g);
            acc[0][mt] = wmma_bf16(aw0, bx, acc[0][mt]);
            acc[1][mt] = wmma_bf16(aw1, bx, acc[1][mt]);
        }
    }

    // C^T tile element: vgpr r -> channel n = nb + r, lane -> row m.
    #pragma unroll
    for (int nt = 0; nt < 2; ++nt) {
        const int nb = nblk + wave * 32 + nt * 16 + 8 * g;
        F8 bv;
        bv.q[0] = *reinterpret_cast<const float4*>(bias + nb);
        bv.q[1] = *reinterpret_cast<const float4*>(bias + nb + 4);
        #pragma unroll
        for (int mt = 0; mt < 4; ++mt) {
            int m = mblk + mt * 16 + r16;
            size_t base = (size_t)m * N + nb;
            F8 o;
            #pragma unroll
            for (int r = 0; r < 8; ++r) o.f[r] = (acc[nt][mt][r] + bv.f[r]) * scale;
            if (resid) {
                F8 rr;
                rr.q[0] = *reinterpret_cast<const float4*>(resid + base);
                rr.q[1] = *reinterpret_cast<const float4*>(resid + base + 4);
                #pragma unroll
                for (int r = 0; r < 8; ++r) o.f[r] += rr.f[r];
            }
            if (outf) {
                *reinterpret_cast<float4*>(outf + base) = o.q[0];
                *reinterpret_cast<float4*>(outf + base + 4) = o.q[1];
            } else if (transpose_out) {
                #pragma unroll
                for (int r = 0; r < 8; ++r)
                    outb[(size_t)(nb + r) * M + m] = f2bf(o.f[r]);
            } else {
                BF8PK pk;
                #pragma unroll
                for (int r = 0; r < 8; ++r) pk.b[r] = f2bf(o.f[r]);
                *reinterpret_cast<uint4*>(outb + base) = pk.q;
            }
        }
    }
}

// ---------------------------------------------------------------------------
// Flash attention. grid = (64 q-blocks, 32 b*h), 128 threads (4 waves).
// Computes S^T = K * Q^T (16 keys x 16 queries tiles). Each lane owns one
// query (col), so softmax reductions are a single shfl_xor(.,16) and the
// running max/sum are per-lane scalars. The S^T accumulator layout is
// identical to the B-operand layout of O^T = V^T * P^T, so P stays in VGPRs.
// Q pre-scaled by log2e/sqrt(dk); Vt is (512 ch x 4096 rows) bf16.
// ---------------------------------------------------------------------------
__global__ __launch_bounds__(128) void attn_kernel(
    const __bf16* __restrict__ Q, const __bf16* __restrict__ Kb,
    const __bf16* __restrict__ Vt, __bf16* __restrict__ O) {
    __shared__ __bf16 Kl[128 * 64];   // [key][k]
    __shared__ __bf16 Vl[64 * 128];   // [dim][key]  (transposed V tile)

    const int bh = blockIdx.y, b = bh >> 3, h = bh & 7;
    const int lane = threadIdx.x & 31, wave = threadIdx.x >> 5;
    const int g = lane >> 4, r16 = lane & 15;
    const int qbase = blockIdx.x * 64 + wave * 16;

    // Q fragments (B operand: lane = query col, contiguous in k), loaded once
    const __bf16* qptr = Q + (size_t)(b * 4096 + qbase + r16) * 512 + h * 64;
    v16bf qb0 = ldfrag(qptr, g);
    v16bf qb1 = ldfrag(qptr + 32, g);

    v8f oa[4] = {};          // O^T tiles: [dim-tile] (16 dims x 16 queries)
    float rmax = -3.0e38f, rsum = 0.0f;

    for (int kt = 0; kt < 8; ++kt) {
        __syncthreads();
        {   // stage K tile [128 key][64 k] and V^T tile [64 dim][128 key]
            int t = threadIdx.x;
            size_t srcK = (size_t)(b * 1024 + kt * 128 + t) * 512 + h * 64;
            const uint4* sk = reinterpret_cast<const uint4*>(Kb + srcK);
            uint4* dk = reinterpret_cast<uint4*>(Kl + t * 64);
            int d = t >> 1, half = (t & 1) * 64;
            size_t srcV = (size_t)(h * 64 + d) * 4096 + b * 1024 + kt * 128 + half;
            const uint4* sv = reinterpret_cast<const uint4*>(Vt + srcV);
            uint4* dv = reinterpret_cast<uint4*>(Vl + d * 128 + half);
            #pragma unroll
            for (int u = 0; u < 8; ++u) dk[u] = sk[u];
            #pragma unroll
            for (int u = 0; u < 8; ++u) dv[u] = sv[u];
        }
        __syncthreads();

        // S^T tiles: 8 x (16 keys x 16 queries), k = 64 in two 32-chunks
        v8f st[8];
        #pragma unroll
        for (int t = 0; t < 8; ++t) {
            v16bf ka0 = ldfrag(Kl + (t * 16 + r16) * 64, g);
            v16bf ka1 = ldfrag(Kl + (t * 16 + r16) * 64 + 32, g);
            v8f a = {};
            a = wmma_bf16(ka0, qb0, a);
            a = wmma_bf16(ka1, qb1, a);
            st[t] = a;
        }

        // online softmax; every element in this lane belongs to one query
        float cm = st[0][0];
        #pragma unroll
        for (int t = 0; t < 8; ++t)
            #pragma unroll
            for (int r = 0; r < 8; ++r) cm = fmaxf(cm, st[t][r]);
        cm = fmaxf(cm, __shfl_xor(cm, 16, 32));
        float nm = fmaxf(rmax, cm);
        float alpha = exp2f(rmax - nm);
        rmax = nm;

        // P^T = exp2(S^T - max), converted straight into B-operand frags:
        // frag kc covers keys kc*32..+31 = S^T tiles 2kc (j<8) and 2kc+1 (j>=8)
        v16bf pb[4];
        float psum = 0.0f;
        #pragma unroll
        for (int t = 0; t < 8; ++t)
            #pragma unroll
            for (int r = 0; r < 8; ++r) {
                float p = exp2f(st[t][r] - nm);
                psum += p;
                if (t & 1) pb[t >> 1][8 + r] = f2bf(p);
                else       pb[t >> 1][r]     = f2bf(p);
            }
        psum += __shfl_xor(psum, 16, 32);
        rsum = rsum * alpha + psum;

        #pragma unroll
        for (int dt = 0; dt < 4; ++dt)
            #pragma unroll
            for (int r = 0; r < 8; ++r) oa[dt][r] *= alpha;

        // O^T += V^T * P^T : A-op = V^T rows (16 dims x 32 keys)
        #pragma unroll
        for (int kc = 0; kc < 4; ++kc)
            #pragma unroll
            for (int dt = 0; dt < 4; ++dt) {
                v16bf va = ldfrag(Vl + (dt * 16 + r16) * 128 + kc * 32, g);
                oa[dt] = wmma_bf16(va, pb[kc], oa[dt]);
            }
    }

    // normalize; lane writes its query's 8 consecutive dims per tile (16B)
    float inv = 1.0f / rsum;
    size_t obase = (size_t)(b * 4096 + qbase + r16) * 512 + h * 64 + 8 * g;
    #pragma unroll
    for (int dt = 0; dt < 4; ++dt) {
        BF8PK pk;
        #pragma unroll
        for (int r = 0; r < 8; ++r) pk.b[r] = f2bf(oa[dt][r] * inv);
        *reinterpret_cast<uint4*>(O + obase + dt * 16) = pk.q;
    }
}

// ---------------------------------------------------------------------------
extern "C" void kernel_launch(void* const* d_in, const int* in_sizes, int n_in,
                              void* d_out, int out_size, void* d_ws, size_t ws_size,
                              hipStream_t stream) {
    (void)in_sizes; (void)n_in; (void)out_size; (void)ws_size;
    const float* x   = (const float*)d_in[0];
    const float* w_q = (const float*)d_in[1];
    const float* b_q = (const float*)d_in[2];
    const float* w_k = (const float*)d_in[3];
    const float* b_k = (const float*)d_in[4];
    const float* w_v = (const float*)d_in[5];
    const float* b_v = (const float*)d_in[6];
    const float* w_o = (const float*)d_in[7];
    const float* b_o = (const float*)d_in[8];
    float* out = (float*)d_out;

    const int M  = 4 * 64 * 64;   // 16384 full-res pixels
    const int Mp = 4 * 32 * 32;   // 4096  pooled pixels
    const int C  = 512;

    size_t off = 0;
    auto take = [&](size_t bytes) { size_t o = off; off = (off + bytes + 255) & ~(size_t)255; return o; };
    char* ws = (char*)d_ws;
    __bf16* xbf  = (__bf16*)(ws + take((size_t)M  * C * 2));
    __bf16* xpbf = (__bf16*)(ws + take((size_t)Mp * C * 2));
    __bf16* wqt  = (__bf16*)(ws + take((size_t)C * C * 2));   // W^T (N x K)
    __bf16* wkt  = (__bf16*)(ws + take((size_t)C * C * 2));
    __bf16* wvt  = (__bf16*)(ws + take((size_t)C * C * 2));
    __bf16* wot  = (__bf16*)(ws + take((size_t)C * C * 2));
    __bf16* Qbf  = (__bf16*)(ws + take((size_t)M  * C * 2));  // (M x C)
    __bf16* Kbf  = (__bf16*)(ws + take((size_t)Mp * C * 2));  // (Mp x C)
    __bf16* Vtb  = (__bf16*)(ws + take((size_t)C * Mp * 2));  // (C x Mp) transposed
    __bf16* AObf = (__bf16*)(ws + take((size_t)M  * C * 2));  // (M x C)

    // 1) precision conversion + pooling (+ weight transpose)
    cvt_bf16_kernel<<<(M * C + 255) / 256, 256, 0, stream>>>(x, xbf, M * C);
    pool_cvt_kernel<<<(Mp * C + 255) / 256, 256, 0, stream>>>(x, xpbf);
    cvt_t_bf16_kernel<<<(C * C + 255) / 256, 256, 0, stream>>>(w_q, wqt, C, C);
    cvt_t_bf16_kernel<<<(C * C + 255) / 256, 256, 0, stream>>>(w_k, wkt, C, C);
    cvt_t_bf16_kernel<<<(C * C + 255) / 256, 256, 0, stream>>>(w_v, wvt, C, C);
    cvt_t_bf16_kernel<<<(C * C + 255) / 256, 256, 0, stream>>>(w_o, wot, C, C);

    // 2) projections. Q gets log2e / sqrt(d_k) folded in (softmax uses exp2).
    const float qscale = 1.4426950408889634f / 8.0f;
    gemm_bf16_kernel<<<dim3(C / 128, M / 64), 128, 0, stream>>>(
        xbf, wqt, b_q, nullptr, Qbf, nullptr, M, C, C, qscale, 0);
    gemm_bf16_kernel<<<dim3(C / 128, Mp / 64), 128, 0, stream>>>(
        xpbf, wkt, b_k, nullptr, Kbf, nullptr, Mp, C, C, 1.0f, 0);
    gemm_bf16_kernel<<<dim3(C / 128, Mp / 64), 128, 0, stream>>>(
        xpbf, wvt, b_v, nullptr, Vtb, nullptr, Mp, C, C, 1.0f, 1);

    // 3) flash attention: 64 q-blocks x 32 (b,h)
    attn_kernel<<<dim3(64, 32), 128, 0, stream>>>(Qbf, Kbf, Vtb, AObf);

    // 4) output projection + bias + residual -> fp32 d_out
    gemm_bf16_kernel<<<dim3(C / 128, M / 64), 128, 0, stream>>>(
        AObf, wot, b_o, x, nullptr, out, M, C, C, 1.0f, 0);
}
